// GlobalLocalAttention_56573309223860
// MI455X (gfx1250) — matrix operand order repaired and checked by reading, not compiled, not run on hardware
//
#include <hip/hip_runtime.h>

typedef __attribute__((ext_vector_type(16))) __bf16 v16bf;
typedef __attribute__((ext_vector_type(8)))  float  v8f;

#define S_LEN      2048
#define D_DIM      128
#define N_GLOBAL   16
#define HALF_WIN   256
#define QROWS      64      // query rows per block (4 waves x 16)
#define KTILE      32      // keys per KV tile
#define KPAD       (D_DIM + 4)
#define VPAD       (KTILE + 2)

__device__ __forceinline__ __bf16 to_bf16(float f) { return (__bf16)f; }

__device__ __forceinline__ v8f vzero8() {
    v8f z;
#pragma unroll
    for (int i = 0; i < 8; ++i) z[i] = 0.0f;
    return z;
}

__global__ __launch_bounds__(128)
void GlobalLocalAttention_gla_kernel(const float* __restrict__ q,
                                     const float* __restrict__ k,
                                     const float* __restrict__ v,
                                     float* __restrict__ out)
{
    const int nqb = S_LEN / QROWS;              // 32 query blocks per head
    const int bh  = blockIdx.x / nqb;
    const int qb0 = (blockIdx.x % nqb) * QROWS;

    const float* Q = q   + (size_t)bh * S_LEN * D_DIM;
    const float* K = k   + (size_t)bh * S_LEN * D_DIM;
    const float* V = v   + (size_t)bh * S_LEN * D_DIM;
    float*       O = out + (size_t)bh * S_LEN * D_DIM;

    const int tid  = threadIdx.x;
    const int wave = tid >> 5;
    const int lane = tid & 31;
    const int m16  = lane & 15;
    const int half = lane >> 4;

    // Raw f32 staging (single buffer: its lifetime ends at the convert pass,
    // before compute, so async fill of tile t+1 overlaps WMMA of tile t).
    __shared__ float4 kraw[(KTILE * D_DIM) / 4];     // 16 KB, contiguous tile
    __shared__ float4 vraw[(KTILE * D_DIM) / 4];     // 16 KB
    __shared__ __bf16 k_lds[KTILE][KPAD];            // [key][feat]
    __shared__ __bf16 vt_lds[D_DIM][VPAD];           // [feat][key] (transposed)
    __shared__ __bf16 p_lds[4][16][VPAD];            // per-wave P relayout scratch

    // LDS byte offsets of the raw buffers (flat addr low 32 bits = DS offset)
    const unsigned kraw_off = (unsigned)(uintptr_t)(void*)&kraw[0];
    const unsigned vraw_off = (unsigned)(uintptr_t)(void*)&vraw[0];

    // A-fragment K map (16-bit 16x32): lanes0-15 -> K groups {0-7,16-23},
    // lanes16-31 -> {8-15,24-31}; 2 values per VGPR.
    int kmapA[8];
#pragma unroll
    for (int vv = 0; vv < 8; ++vv)
        kmapA[vv] = ((vv < 4) ? (2 * vv) : (16 + 2 * (vv - 4))) + (half ? 8 : 0);
    // B-fragment K map (32x16): same as A: lanes0-15 -> K {0-7,16-23}, lanes16-31 -> {8-15,24-31}.
    int kmapB[8];
#pragma unroll
    for (int vv = 0; vv < 8; ++vv)
        kmapB[vv] = ((vv < 4) ? (2 * vv) : (16 + 2 * (vv - 4))) + (half ? 8 : 0);

    // ---- KV tile range: union of windows for rows [qb0, qb0+63], plus tile 0
    int tlo = (qb0 > HALF_WIN) ? (qb0 - HALF_WIN) / KTILE : 0;
    int thi = qb0 + QROWS + HALF_WIN;
    if (thi > S_LEN) thi = S_LEN;
    thi = (thi + KTILE - 1) / KTILE;
    if (qb0 == 0) thi = S_LEN / KTILE;           // global rows attend everything
    const int ntiles = (thi - tlo) + (tlo > 0 ? 1 : 0);

    // tile 0 (global columns) is always processed first so row-max is finite
    auto tile_of = [&](int i) {
        return (tlo > 0) ? ((i == 0) ? 0 : (tlo + i - 1)) : (tlo + i);
    };

    // Async stage of one KV tile: two contiguous 16 KB blocks, B128 per lane.
    // GVS mode: per-lane u32 offset + uniform SGPR base.  16 asyncs per wave.
    auto stage_tile = [&](int jt) {
        const unsigned goff0 = (unsigned)jt * (unsigned)(D_DIM * sizeof(float));
#pragma unroll
        for (int it = 0; it < 8; ++it) {
            const unsigned b16 = ((unsigned)tid + (unsigned)it * 128u) * 16u;
            const unsigned go  = goff0 + b16;
            const unsigned lk  = kraw_off + b16;
            const unsigned lv  = vraw_off + b16;
            asm volatile("global_load_async_to_lds_b128 %0, %1, %2"
                         :: "v"(lk), "v"(go), "s"(K) : "memory");
            asm volatile("global_load_async_to_lds_b128 %0, %1, %2"
                         :: "v"(lv), "v"(go), "s"(V) : "memory");
        }
    };

    // Kick off tile 0's DMA before touching Q (overlaps with Q global loads).
    stage_tile(tile_of(0) * KTILE);

    // ---- Load Q tile into 4 A fragments (rows q0w..q0w+15, D=128 = 4 x K32)
    const int q0w = qb0 + wave * 16;
    v16bf a_q[4];
    {
        const float* qrow = Q + (size_t)(q0w + m16) * D_DIM;
#pragma unroll
        for (int c = 0; c < 4; ++c) {
#pragma unroll
            for (int vv = 0; vv < 8; ++vv) {
                const int k0 = c * 32 + kmapA[vv];
                a_q[c][2 * vv]     = to_bf16(qrow[k0]);
                a_q[c][2 * vv + 1] = to_bf16(qrow[k0 + 1]);
            }
        }
    }

    v8f acc[8];
#pragma unroll
    for (int o = 0; o < 8; ++o) acc[o] = vzero8();
    float mrow[8], lrow[8];
#pragma unroll
    for (int r = 0; r < 8; ++r) { mrow[r] = -INFINITY; lrow[r] = 0.0f; }

    const float scale = 0.08838834764831843f;    // 1/sqrt(128)

    for (int tt = 0; tt < ntiles; ++tt) {
        const int jt = tile_of(tt) * KTILE;

        // Own wave's async loads landed; barrier covers the other waves'.
        asm volatile("s_wait_asynccnt 0x0" ::: "memory");
        __syncthreads();

        // ---- convert raw f32 tile -> bf16 K (row-major) and V (transposed)
#pragma unroll
        for (int it = 0; it < 8; ++it) {
            const int idx = tid + it * 128;      // float4 chunk index
            const int key = idx >> 5;            // 0..31
            const int f4  = (idx & 31) << 2;     // feature*4
            const float4 kk4 = kraw[idx];
            k_lds[key][f4 + 0] = to_bf16(kk4.x);
            k_lds[key][f4 + 1] = to_bf16(kk4.y);
            k_lds[key][f4 + 2] = to_bf16(kk4.z);
            k_lds[key][f4 + 3] = to_bf16(kk4.w);
            const float4 vv4 = vraw[idx];
            vt_lds[f4 + 0][key] = to_bf16(vv4.x);
            vt_lds[f4 + 1][key] = to_bf16(vv4.y);
            vt_lds[f4 + 2][key] = to_bf16(vv4.z);
            vt_lds[f4 + 3][key] = to_bf16(vv4.w);
        }
        __syncthreads();   // bf16 ready; raw buffer now dead -> safe to refill

        // Overlap next tile's HBM fetch with this tile's WMMA compute.
        if (tt + 1 < ntiles)
            stage_tile(tile_of(tt + 1) * KTILE);

        // ---- S = Q K^T for two 16-key sub-tiles (8 WMMAs)
        v8f s0 = vzero8(), s1 = vzero8();
#pragma unroll
        for (int c = 0; c < 4; ++c) {
            v16bf b0, b1;
#pragma unroll
            for (int vv = 0; vv < 8; ++vv) {
                const int kf = c * 32 + kmapB[vv];
                b0[2 * vv]     = k_lds[m16][kf];
                b0[2 * vv + 1] = k_lds[m16][kf + 1];
                b1[2 * vv]     = k_lds[m16 + 16][kf];
                b1[2 * vv + 1] = k_lds[m16 + 16][kf + 1];
            }
            s0 = __builtin_amdgcn_wmma_f32_16x16x32_bf16(false, a_q[c], false, b0,
                                                         (short)0, s0, false, false);
            s1 = __builtin_amdgcn_wmma_f32_16x16x32_bf16(false, a_q[c], false, b1,
                                                         (short)0, s1, false, false);
        }

        // ---- mask + online softmax (C layout: elem r -> row r+8*half, col lane%16)
        const int j0 = jt + m16;
        const int j1 = jt + 16 + m16;
        float p0[8], p1[8];
#pragma unroll
        for (int r = 0; r < 8; ++r) {
            const int i = q0w + r + 8 * half;
            float v0 = s0[r] * scale;
            float v1 = s1[r] * scale;
            const bool a0 = ((j0 >= i - HALF_WIN) && (j0 <= i + HALF_WIN)) ||
                            (j0 < N_GLOBAL) || (i < N_GLOBAL);
            const bool a1 = ((j1 >= i - HALF_WIN) && (j1 <= i + HALF_WIN)) ||
                            (j1 < N_GLOBAL) || (i < N_GLOBAL);
            v0 = a0 ? v0 : -INFINITY;
            v1 = a1 ? v1 : -INFINITY;
            float mx = fmaxf(v0, v1);
            mx = fmaxf(mx, __shfl_xor(mx, 1, 32));
            mx = fmaxf(mx, __shfl_xor(mx, 2, 32));
            mx = fmaxf(mx, __shfl_xor(mx, 4, 32));
            mx = fmaxf(mx, __shfl_xor(mx, 8, 32));
            const float mnew = fmaxf(mrow[r], mx);
            const float sc   = __expf(mrow[r] - mnew);   // 0 on first real tile
            mrow[r] = mnew;
            const float e0 = __expf(v0 - mnew);
            const float e1 = __expf(v1 - mnew);
            float rs = e0 + e1;
            rs += __shfl_xor(rs, 1, 32);
            rs += __shfl_xor(rs, 2, 32);
            rs += __shfl_xor(rs, 4, 32);
            rs += __shfl_xor(rs, 8, 32);
            lrow[r] = lrow[r] * sc + rs;
            p0[r] = e0;
            p1[r] = e1;
#pragma unroll
            for (int o = 0; o < 8; ++o) acc[o][r] *= sc;
        }

        // ---- relayout P (C layout -> A layout) through per-wave LDS
#pragma unroll
        for (int r = 0; r < 8; ++r) {
            const int row = r + 8 * half;
            p_lds[wave][row][m16]      = to_bf16(p0[r]);
            p_lds[wave][row][16 + m16] = to_bf16(p1[r]);
        }
        asm volatile("s_wait_dscnt 0" ::: "memory");

        v16bf a_p;
#pragma unroll
        for (int vv = 0; vv < 8; ++vv) {
            const int k0 = kmapA[vv];
            a_p[2 * vv]     = p_lds[wave][m16][k0];
            a_p[2 * vv + 1] = p_lds[wave][m16][k0 + 1];
        }

        // ---- O += P V (8 WMMAs over the 128 output features)
#pragma unroll
        for (int o = 0; o < 8; ++o) {
            v16bf b_v;
#pragma unroll
            for (int vv = 0; vv < 8; ++vv) {
                const int k0 = kmapB[vv];
                b_v[2 * vv]     = vt_lds[o * 16 + m16][k0];
                b_v[2 * vv + 1] = vt_lds[o * 16 + m16][k0 + 1];
            }
            acc[o] = __builtin_amdgcn_wmma_f32_16x16x32_bf16(false, a_p, false, b_v,
                                                             (short)0, acc[o], false, false);
        }
    }

    // ---- epilogue: normalize and store fp32
#pragma unroll
    for (int r = 0; r < 8; ++r) {
        const float inv = 1.0f / lrow[r];
        const int row = q0w + r + 8 * half;
#pragma unroll
        for (int o = 0; o < 8; ++o) {
            O[(size_t)row * D_DIM + o * 16 + m16] = acc[o][r] * inv;
        }
    }
}

extern "C" void kernel_launch(void* const* d_in, const int* in_sizes, int n_in,
                              void* d_out, int out_size, void* d_ws, size_t ws_size,
                              hipStream_t stream) {
    (void)n_in; (void)out_size; (void)d_ws; (void)ws_size;
    const float* q = (const float*)d_in[0];
    const float* k = (const float*)d_in[1];
    const float* v = (const float*)d_in[2];
    float* out = (float*)d_out;

    const int BH = in_sizes[0] / (S_LEN * D_DIM);       // B*H = 32
    const int blocks = BH * (S_LEN / QROWS);            // 1024
    GlobalLocalAttention_gla_kernel<<<blocks, 128, 0, stream>>>(q, k, v, out);
}
